// DFM_MeshCNN_37641093382407
// MI455X (gfx1250) — compile-verified, matrix operand order checked by reading, not run on hardware
//
#include <hip/hip_runtime.h>
#include <hip/hip_bf16.h>
#include <stdint.h>

typedef _Float16 v8h  __attribute__((ext_vector_type(8)));
typedef _Float16 v16h __attribute__((ext_vector_type(16)));
typedef float    v8f  __attribute__((ext_vector_type(8)));

#define EPSV 1e-5f
#define NB    64
#define E_PER 20000
#define NE    (NB * E_PER)     // 1,280,000
#define KP1   512
#define KP2   256
#define N1    (NB * KP1)       // 32768
#define N2    (NB * KP2)       // 16384
#define GSTAT 11
#define FUSED (256 + GSTAT)    // 267

// ---------------------------------------------------------------------------
// Conv1 (Cin=5, Cout=64): VALU kernel. SCORE_ONLY computes ||relu(bn(.))||^2
// for top-k; otherwise recomputes features for the kept edges into f16.
// ---------------------------------------------------------------------------
template <bool SCORE_ONLY>
__global__ void conv1_kernel(const float* __restrict__ x, const int* __restrict__ e2e,
                             const float* __restrict__ W, const float* __restrict__ bi,
                             const float* __restrict__ bg, const float* __restrict__ bb,
                             const float* __restrict__ bm, const float* __restrict__ bv,
                             const int* __restrict__ kept, float* __restrict__ scores,
                             _Float16* __restrict__ xout, int n) {
  int i = blockIdx.x * blockDim.x + threadIdx.x;
  if (i >= n) return;
  int e = SCORE_ONLY ? i : kept[i];
  float in[5][5];
#pragma unroll
  for (int c = 0; c < 5; ++c) in[0][c] = x[(size_t)e * 5 + c];
#pragma unroll
  for (int p = 1; p < 5; ++p) {
    int nb = e2e[(size_t)e * 4 + (p - 1)];
#pragma unroll
    for (int c = 0; c < 5; ++c) in[p][c] = x[(size_t)nb * 5 + c];
  }
  float ss = 0.f;
  for (int o = 0; o < 64; ++o) {
    float acc = bi[o];
#pragma unroll
    for (int c = 0; c < 5; ++c)
#pragma unroll
      for (int p = 0; p < 5; ++p)
        acc = fmaf(in[p][c], W[(o * 5 + c) * 5 + p], acc);
    float val = (acc - bm[o]) * (bg[o] * rsqrtf(bv[o] + EPSV)) + bb[o];
    val = fmaxf(val, 0.f);
    if (SCORE_ONLY) ss += val * val;
    else            xout[(size_t)i * 64 + o] = (_Float16)val;
  }
  if (SCORE_ONLY) scores[i] = ss;
}

// ---------------------------------------------------------------------------
// Top-k per graph: radix/bit binary-search on order-preserving u32 keys.
// Selects exactly k indices per graph into kept[g*k .. g*k+k-1] (set-equal
// to lax.top_k; downstream segment-sum is permutation invariant).
// ---------------------------------------------------------------------------
__device__ __forceinline__ unsigned fkey(float f) {
  unsigned u = __float_as_uint(f);
  return (u & 0x80000000u) ? ~u : (u | 0x80000000u);
}

__global__ void topk_kernel(const float* __restrict__ scores, int* __restrict__ kept,
                            int per, int k) {
  __shared__ unsigned s_cnt;
  __shared__ int s_hi, s_tie;
  int g = blockIdx.x, tid = threadIdx.x, bs = blockDim.x;
  const float* sg = scores + (size_t)g * per;

  unsigned prefix = 0;
  for (int b = 31; b >= 0; --b) {
    unsigned trial = prefix | (1u << b);
    if (tid == 0) s_cnt = 0;
    __syncthreads();
    unsigned local = 0;
    for (int i = tid; i < per; i += bs) local += (fkey(sg[i]) >= trial) ? 1u : 0u;
    atomicAdd(&s_cnt, local);
    __syncthreads();
    if (s_cnt >= (unsigned)k) prefix = trial;
    __syncthreads();
  }
  unsigned T = prefix;
  if (tid == 0) s_cnt = 0;
  __syncthreads();
  unsigned local = 0;
  for (int i = tid; i < per; i += bs) local += (fkey(sg[i]) > T) ? 1u : 0u;
  atomicAdd(&s_cnt, local);
  __syncthreads();
  int cg = (int)s_cnt;        // strictly greater than threshold
  int need = k - cg;          // ties to take
  if (tid == 0) { s_hi = 0; s_tie = 0; }
  __syncthreads();
  for (int i = tid; i < per; i += bs) {
    unsigned key = fkey(sg[i]);
    if (key > T) {
      int pos = atomicAdd(&s_hi, 1);
      kept[g * k + pos] = g * per + i;
    } else if (key == T) {
      int pos = atomicAdd(&s_tie, 1);
      if (pos < need) kept[g * k + cg + pos] = g * per + i;
    }
  }
}

// ---------------------------------------------------------------------------
// Pool bookkeeping
// ---------------------------------------------------------------------------
__global__ void fill_i32(int* __restrict__ p, int val, int n) {
  int i = blockIdx.x * blockDim.x + threadIdx.x;
  if (i < n) p[i] = val;
}
__global__ void scatter_new(const int* __restrict__ kept, int* __restrict__ old2new, int n_new) {
  int i = blockIdx.x * blockDim.x + threadIdx.x;
  if (i < n_new) old2new[kept[i]] = i;
}
__global__ void remap_e2e(const int* __restrict__ kept, const int* __restrict__ e2e_old,
                          const int* __restrict__ old2new, int* __restrict__ e2e_new, int n_new) {
  int i = blockIdx.x * blockDim.x + threadIdx.x;
  if (i >= n_new * 4) return;
  int r = i >> 2, j = i & 3;
  int nb = e2e_old[(size_t)kept[r] * 4 + j];
  int v = old2new[nb];
  e2e_new[i] = (v < 0) ? r : v;
}
__global__ void compact_f32_to_f16(const float* __restrict__ x, const int* __restrict__ kept,
                                   _Float16* __restrict__ out, int cols, int n_new) {
  int idx = blockIdx.x * blockDim.x + threadIdx.x;
  if (idx >= n_new * cols) return;
  int r = idx / cols, c = idx % cols;
  out[idx] = (_Float16)x[(size_t)kept[r] * cols + c];
}

// ---------------------------------------------------------------------------
// Pack conv weights (o,c,p) -> WMMA B-fragment layout (f16):
// Bp[((kk*NT + nt)*32 + lane)*16 + elem], one contiguous 32B per lane per kk.
// k = p*CIN + c maps to B[k][n] = W[n][c][p].
// ---------------------------------------------------------------------------
template <int CIN, int COUT>
__global__ void pack_b(const float* __restrict__ W, _Float16* __restrict__ Bp) {
  constexpr int NKT = (5 * CIN) / 32;
  constexpr int NT  = COUT / 16;
  int tid = blockIdx.x * blockDim.x + threadIdx.x;
  if (tid >= NKT * NT * 32 * 16) return;
  int elem = tid & 15;
  int lane = (tid >> 4) & 31;
  int nt   = (tid >> 9) % NT;
  int kk   = tid / (NT * 512);
  int hi   = lane >> 4;
  int k = kk * 32 + ((elem < 8) ? (hi * 8 + elem) : (16 + hi * 8 + (elem - 8)));
  int n = nt * 16 + (lane & 15);
  int p = k / CIN, c = k % CIN;
  Bp[tid] = (_Float16)W[((size_t)n * CIN + c) * 5 + p];
}

// ---------------------------------------------------------------------------
// WMMA mesh-conv GEMM: out[e,o] = relu(bn( sum_k A[e,k]*B[k,o] + bias )),
// A[e, p*CIN+c] = x[(p==0 ? e : e2e[e][p-1])][c]  (gathered directly from
// global: each lane's 8-half K-run is contiguous inside one feature row).
// One wave = MTILES (4) independent 16x16 output tiles sharing one B
// fragment per K-step: 4x less B traffic + 4 independent wmma chains.
// ---------------------------------------------------------------------------
#define MTILES 4

template <int CIN>
__device__ __forceinline__ v8h load_a8(const _Float16* __restrict__ xin,
                                       const int* __restrict__ e2e, int e, int k) {
  int p = k / CIN;
  int c = k % CIN;
  int src = (p == 0) ? e : e2e[(size_t)e * 4 + (p - 1)];
  return *(const v8h*)(xin + (size_t)src * CIN + c);
}

template <int CIN, int COUT>
__global__ __launch_bounds__(128) void conv_wmma(
    const _Float16* __restrict__ xin, const int* __restrict__ e2e,
    const _Float16* __restrict__ Bp, const float* __restrict__ bias,
    const float* __restrict__ bg, const float* __restrict__ bb,
    const float* __restrict__ bm, const float* __restrict__ bv,
    float* __restrict__ out_f32) {
  constexpr int NKT = (5 * CIN) / 32;
  constexpr int NT  = COUT / 16;
  int wave  = threadIdx.x >> 5;
  int lane  = threadIdx.x & 31;
  int ntile = blockIdx.y * 4 + wave;
  int mbase = blockIdx.x * (16 * MTILES);
  int hi    = (lane >> 4) & 1;
  int mlane = lane & 15;

  v8f acc[MTILES];
#pragma unroll
  for (int t = 0; t < MTILES; ++t)
    acc[t] = (v8f){0.f, 0.f, 0.f, 0.f, 0.f, 0.f, 0.f, 0.f};

#pragma unroll
  for (int kk = 0; kk < NKT; ++kk) {
    int k0 = kk * 32 + hi * 8;
    // B fragment: loaded once, reused by all MTILES wmma's this K-step.
    const _Float16* bp = Bp + (((size_t)kk * NT + ntile) * 32 + lane) * 16;
    v8h blo = *(const v8h*)bp;
    v8h bhi = *(const v8h*)(bp + 8);
    v16h b = __builtin_shufflevector(blo, bhi, 0, 1, 2, 3, 4, 5, 6, 7,
                                     8, 9, 10, 11, 12, 13, 14, 15);
#pragma unroll
    for (int t = 0; t < MTILES; ++t) {
      int em = mbase + t * 16 + mlane;
      v8h alo = load_a8<CIN>(xin, e2e, em, k0);
      v8h ahi = load_a8<CIN>(xin, e2e, em, k0 + 16);
      v16h a = __builtin_shufflevector(alo, ahi, 0, 1, 2, 3, 4, 5, 6, 7,
                                       8, 9, 10, 11, 12, 13, 14, 15);
      acc[t] = __builtin_amdgcn_wmma_f32_16x16x32_f16(false, a, false, b,
                                                      (short)0, acc[t], false, false);
    }
  }
  // Epilogue: D layout -> lane holds col = lane&15, rows r + hi*8.
  int col = ntile * 16 + mlane;
  float bi = bias[col];
  float sc = bg[col] * rsqrtf(bv[col] + EPSV);
  float mm = bm[col], be = bb[col];
#pragma unroll
  for (int t = 0; t < MTILES; ++t) {
#pragma unroll
    for (int r = 0; r < 8; ++r) {
      int row = mbase + t * 16 + r + hi * 8;
      float v = (acc[t][r] + bi - mm) * sc + be;
      out_f32[(size_t)row * COUT + col] = fmaxf(v, 0.f);
    }
  }
}

// ---------------------------------------------------------------------------
// Row ||.||^2 (one wave per row) for pool-2 scores.
// ---------------------------------------------------------------------------
__global__ void row_sumsq(const float* __restrict__ x, float* __restrict__ scores,
                          int rows, int cols) {
  int wave = (blockIdx.x * blockDim.x + threadIdx.x) >> 5;
  int lane = threadIdx.x & 31;
  if (wave >= rows) return;
  float s = 0.f;
  for (int c = lane; c < cols; c += 32) {
    float v = x[(size_t)wave * cols + c];
    s += v * v;
  }
#pragma unroll
  for (int off = 16; off; off >>= 1) s += __shfl_xor(s, off, 32);
  if (lane == 0) scores[wave] = s;
}

// ---------------------------------------------------------------------------
// Segment mean over the (implicitly graph-grouped) conv3 rows + fuse stats.
// ---------------------------------------------------------------------------
__global__ void pool_mean_fuse(const float* __restrict__ x3, const float* __restrict__ gstats,
                               float* __restrict__ fused) {
  int g = blockIdx.x, c = threadIdx.x;  // 256 threads
  float s = 0.f;
  for (int r = 0; r < KP2; ++r) s += x3[((size_t)(g * KP2 + r)) * 256 + c];
  fused[g * FUSED + c] = s * (1.f / (float)KP2);
  if (c < GSTAT) fused[g * FUSED + 256 + c] = gstats[g * GSTAT + c];
}

// ---------------------------------------------------------------------------
// Small dense layer (the MLP heads are ~MFLOP scale).
// ---------------------------------------------------------------------------
__global__ void dense(const float* __restrict__ in, const float* __restrict__ W,
                      const float* __restrict__ b, float* __restrict__ out,
                      int M, int Din, int Dout, int relu) {
  int idx = blockIdx.x * blockDim.x + threadIdx.x;
  if (idx >= M * Dout) return;
  int g = idx / Dout, o = idx % Dout;
  float acc = b[o];
  for (int d = 0; d < Din; ++d) acc = fmaf(in[g * Din + d], W[d * Dout + o], acc);
  if (relu) acc = fmaxf(acc, 0.f);
  out[g * Dout + o] = acc;
}

// ---------------------------------------------------------------------------
// Host-side launch
// ---------------------------------------------------------------------------
static inline char* aup(char* p, size_t a) {
  return (char*)(((uintptr_t)p + (a - 1)) & ~(uintptr_t)(a - 1));
}

extern "C" void kernel_launch(void* const* d_in, const int* in_sizes, int n_in,
                              void* d_out, int out_size, void* d_ws, size_t ws_size,
                              hipStream_t stream) {
  (void)in_sizes; (void)n_in; (void)out_size; (void)ws_size;
  const float* edge_feats = (const float*)d_in[0];
  const int*   e2e        = (const int*)d_in[1];
  // d_in[2] = batch, d_in[3] = edge_index (both implied by our grouping)
  const float* gstats     = (const float*)d_in[4];
  const float* c1_w = (const float*)d_in[5];
  const float* c1_b = (const float*)d_in[6];
  const float* c1_g = (const float*)d_in[7];
  const float* c1_bb = (const float*)d_in[8];
  const float* c1_m = (const float*)d_in[9];
  const float* c1_v = (const float*)d_in[10];
  const float* c2_w = (const float*)d_in[11];
  const float* c2_b = (const float*)d_in[12];
  const float* c2_g = (const float*)d_in[13];
  const float* c2_bb = (const float*)d_in[14];
  const float* c2_m = (const float*)d_in[15];
  const float* c2_v = (const float*)d_in[16];
  const float* c3_w = (const float*)d_in[17];
  const float* c3_b = (const float*)d_in[18];
  const float* c3_g = (const float*)d_in[19];
  const float* c3_bb = (const float*)d_in[20];
  const float* c3_m = (const float*)d_in[21];
  const float* c3_v = (const float*)d_in[22];
  const float* reg_w1 = (const float*)d_in[23]; const float* reg_b1 = (const float*)d_in[24];
  const float* reg_w2 = (const float*)d_in[25]; const float* reg_b2 = (const float*)d_in[26];
  const float* reg_w3 = (const float*)d_in[27]; const float* reg_b3 = (const float*)d_in[28];
  const float* clf_w1 = (const float*)d_in[29]; const float* clf_b1 = (const float*)d_in[30];
  const float* clf_w2 = (const float*)d_in[31]; const float* clf_b2 = (const float*)d_in[32];
  const float* clf_w3 = (const float*)d_in[33]; const float* clf_b3 = (const float*)d_in[34];
  float* out = (float*)d_out;

  // ---- workspace carve-out (all L2-resident: ~55 MB total) ----
  char* p = (char*)d_ws;
  float*     scores1  = (float*)p;          p = aup(p + (size_t)NE * 4, 256);
  int*       kept1    = (int*)p;            p = aup(p + (size_t)N1 * 4, 256);
  int*       old2new1 = (int*)p;            p = aup(p + (size_t)NE * 4, 256);
  int*       e2e2     = (int*)p;            p = aup(p + (size_t)N1 * 16, 256);
  _Float16*  x1h      = (_Float16*)p;       p = aup(p + (size_t)N1 * 64 * 2, 256);
  _Float16*  B2p      = (_Float16*)p;       p = aup(p + (size_t)320 * 128 * 2, 256);
  float*     x2f      = (float*)p;          p = aup(p + (size_t)N1 * 128 * 4, 256);
  float*     scores2  = (float*)p;          p = aup(p + (size_t)N1 * 4, 256);
  int*       kept2    = (int*)p;            p = aup(p + (size_t)N2 * 4, 256);
  int*       old2new2 = (int*)p;            p = aup(p + (size_t)N1 * 4, 256);
  int*       e2e3     = (int*)p;            p = aup(p + (size_t)N2 * 16, 256);
  _Float16*  x2kh     = (_Float16*)p;       p = aup(p + (size_t)N2 * 128 * 2, 256);
  _Float16*  B3p      = (_Float16*)p;       p = aup(p + (size_t)640 * 256 * 2, 256);
  float*     x3f      = (float*)p;          p = aup(p + (size_t)N2 * 256 * 4, 256);
  float*     fused    = (float*)p;          p = aup(p + (size_t)NB * FUSED * 4, 256);
  float*     h1       = (float*)p;          p = aup(p + (size_t)NB * 128 * 4, 256);
  float*     h2       = (float*)p;          p = aup(p + (size_t)NB * 64 * 4, 256);

  // ---- weight pre-pack into WMMA fragment layout (per call; deterministic) ----
  pack_b<64, 128><<<(320 / 32) * (128 / 16) * 512 / 256, 256, 0, stream>>>(c2_w, B2p);
  pack_b<128, 256><<<(640 / 32) * (256 / 16) * 512 / 256, 256, 0, stream>>>(c3_w, B3p);

  // ---- stage 1: conv1 scores only (avoid materializing 327 MB) ----
  conv1_kernel<true><<<(NE + 255) / 256, 256, 0, stream>>>(
      edge_feats, e2e, c1_w, c1_b, c1_g, c1_bb, c1_m, c1_v,
      nullptr, scores1, nullptr, NE);
  topk_kernel<<<NB, 256, 0, stream>>>(scores1, kept1, E_PER, KP1);
  fill_i32<<<(NE + 255) / 256, 256, 0, stream>>>(old2new1, -1, NE);
  scatter_new<<<(N1 + 255) / 256, 256, 0, stream>>>(kept1, old2new1, N1);
  remap_e2e<<<(N1 * 4 + 255) / 256, 256, 0, stream>>>(kept1, e2e, old2new1, e2e2, N1);
  // conv1 recompute on kept edges only -> f16 features for WMMA gathers
  conv1_kernel<false><<<(N1 + 255) / 256, 256, 0, stream>>>(
      edge_feats, e2e, c1_w, c1_b, c1_g, c1_bb, c1_m, c1_v,
      kept1, nullptr, x1h, N1);

  // ---- stage 2: conv2 via WMMA (32768x320 . 320x128) ----
  conv_wmma<64, 128><<<dim3(N1 / (16 * MTILES), 128 / 64), 128, 0, stream>>>(
      x1h, e2e2, B2p, c2_b, c2_g, c2_bb, c2_m, c2_v, x2f);
  row_sumsq<<<(N1 * 32 + 255) / 256, 256, 0, stream>>>(x2f, scores2, N1, 128);
  topk_kernel<<<NB, 256, 0, stream>>>(scores2, kept2, KP1, KP2);
  fill_i32<<<(N1 + 255) / 256, 256, 0, stream>>>(old2new2, -1, N1);
  scatter_new<<<(N2 + 255) / 256, 256, 0, stream>>>(kept2, old2new2, N2);
  remap_e2e<<<(N2 * 4 + 255) / 256, 256, 0, stream>>>(kept2, e2e2, old2new2, e2e3, N2);
  compact_f32_to_f16<<<(N2 * 128 + 255) / 256, 256, 0, stream>>>(x2f, kept2, x2kh, 128, N2);

  // ---- stage 3: conv3 via WMMA (16384x640 . 640x256) ----
  conv_wmma<128, 256><<<dim3(N2 / (16 * MTILES), 256 / 64), 128, 0, stream>>>(
      x2kh, e2e3, B3p, c3_b, c3_g, c3_bb, c3_m, c3_v, x3f);

  // ---- readout: segment mean + fuse graph stats + MLP heads ----
  pool_mean_fuse<<<NB, 256, 0, stream>>>(x3f, gstats, fused);

  dense<<<(NB * 128 + 255) / 256, 256, 0, stream>>>(fused, reg_w1, reg_b1, h1, NB, FUSED, 128, 1);
  dense<<<(NB * 64 + 255) / 256, 256, 0, stream>>>(h1, reg_w2, reg_b2, h2, NB, 128, 64, 1);
  dense<<<1, 64, 0, stream>>>(h2, reg_w3, reg_b3, out, NB, 64, 1, 0);

  dense<<<(NB * 128 + 255) / 256, 256, 0, stream>>>(fused, clf_w1, clf_b1, h1, NB, FUSED, 128, 1);
  dense<<<(NB * 64 + 255) / 256, 256, 0, stream>>>(h1, clf_w2, clf_b2, h2, NB, 128, 64, 1);
  dense<<<(NB * 5 + 255) / 256, 256, 0, stream>>>(h2, clf_w3, clf_b3, out + NB, NB, 64, 5, 0);
}
